// dhgat_45423574123073
// MI455X (gfx1250) — compile-verified, matrix operand order checked by reading, not run on hardware
//
#include <hip/hip_runtime.h>
#include <hip/hip_fp16.h>
#include <math.h>

typedef __attribute__((ext_vector_type(16))) _Float16 v16h;
typedef __attribute__((ext_vector_type(8)))  float    v8f;

#define FDIM 128
#define HID  32
#define GOUT 16
#define NEG_SLOPE 0.2f

__device__ __forceinline__ float leakyf(float x) { return x > 0.f ? x : NEG_SLOPE * x; }
__device__ __forceinline__ float softplusf(float x) {
    return fmaxf(x, 0.f) + log1pf(expf(-fabsf(x)));
}
__device__ __forceinline__ void atomicMaxF(float* addr, float val) {
    unsigned int* ua = (unsigned int*)addr;
    unsigned int old = __float_as_uint(*addr);
    while (__uint_as_float(old) < val) {
        unsigned int assumed = old;
        old = atomicCAS(ua, assumed, __float_as_uint(val));
        if (old == assumed) break;
    }
}

// ---------------------------------------------------------------------------
// Pack a [K=kt*32, Ncols] f32 weight matrix into f16 WMMA B-fragments.
// Fragment f = kt*NT + nt; lane l holds v16h per the ISA 16-bit B layout:
//   lanes 0-15:  N=nt*16+l,  halfs 0-7 = K kb+0..7,   halfs 8-15 = K kb+16..23
//   lanes 16-31: N=nt*16+l', halfs 0-7 = K kb+8..15,  halfs 8-15 = K kb+24..31
// One thread per (frag, lane): total KT*NT*32 threads.
// ---------------------------------------------------------------------------
__global__ void pack_w_kernel(const float* __restrict__ w, _Float16* __restrict__ out,
                              int KT, int NT, int ncols) {
    int tid = blockIdx.x * blockDim.x + threadIdx.x;
    int total = KT * NT * 32;
    if (tid >= total) return;
    int lane = tid & 31;
    int f    = tid >> 5;
    int nt   = f % NT;
    int kt   = f / NT;
    int lm   = lane & 15;
    int koff = (lane < 16) ? 0 : 8;
    int kb   = kt * 32;
    int n    = nt * 16 + lm;
    _Float16* o = out + (size_t)tid * 16;
#pragma unroll
    for (int t = 0; t < 8; ++t) {
        o[t]     = (_Float16)w[(kb + koff + t) * ncols + n];
        o[8 + t] = (_Float16)w[(kb + 16 + koff + t) * ncols + n];
    }
}

// ---------------------------------------------------------------------------
// Fused feature kernel: shr = relu(x@shr_w + shr_b); h = relu(x@spe_w+spe_b)+shr
// One wave per 16-row tile; B fragments pre-packed (v16h vector loads).
// ---------------------------------------------------------------------------
__global__ void feat_kernel(const float* __restrict__ x,
                            const v16h* __restrict__ spe_pk, const float* __restrict__ spe_b,
                            const v16h* __restrict__ shr_pk, const float* __restrict__ shr_b,
                            float* __restrict__ h_out, float* __restrict__ shr_out, int N) {
    int wave  = blockIdx.x * (blockDim.x >> 5) + (threadIdx.x >> 5);
    int lane  = threadIdx.x & 31;
    int tiles = (N + 15) >> 4;
    if (wave >= tiles) return;

    int row0 = wave << 4;
    int lm   = lane & 15;
    int koff = (lane < 16) ? 0 : 8;

    int arow = row0 + lm;
    if (arow >= N) arow = N - 1;
    const float* xrow = x + (size_t)arow * FDIM;

    v8f acc_shr0 = {}; v8f acc_shr1 = {};
    v8f acc_spe0 = {}; v8f acc_spe1 = {};

#pragma unroll
    for (int kt = 0; kt < 4; ++kt) {
        int kb = kt * 32;
        v16h a;
#pragma unroll
        for (int t = 0; t < 8; ++t) {
            a[t]     = (_Float16)xrow[kb + koff + t];
            a[8 + t] = (_Float16)xrow[kb + 16 + koff + t];
        }
        v16h bshr0 = shr_pk[(kt * 2 + 0) * 32 + lane];
        v16h bshr1 = shr_pk[(kt * 2 + 1) * 32 + lane];
        v16h bspe0 = spe_pk[(kt * 2 + 0) * 32 + lane];
        v16h bspe1 = spe_pk[(kt * 2 + 1) * 32 + lane];
        acc_shr0 = __builtin_amdgcn_wmma_f32_16x16x32_f16(false, a, false, bshr0,
                                                          (short)0, acc_shr0, false, false);
        acc_shr1 = __builtin_amdgcn_wmma_f32_16x16x32_f16(false, a, false, bshr1,
                                                          (short)0, acc_shr1, false, false);
        acc_spe0 = __builtin_amdgcn_wmma_f32_16x16x32_f16(false, a, false, bspe0,
                                                          (short)0, acc_spe0, false, false);
        acc_spe1 = __builtin_amdgcn_wmma_f32_16x16x32_f16(false, a, false, bspe1,
                                                          (short)0, acc_spe1, false, false);
    }

    int rbase = row0 + ((lane < 16) ? 0 : 8);
#pragma unroll
    for (int r = 0; r < 8; ++r) {
        int row = rbase + r;
        if (row >= N) continue;
        int c0 = lm;
        float shr0 = fmaxf(acc_shr0[r] + shr_b[c0], 0.f);
        float spe0 = fmaxf(acc_spe0[r] + spe_b[c0], 0.f);
        shr_out[(size_t)row * HID + c0] = shr0;
        h_out  [(size_t)row * HID + c0] = spe0 + shr0;
        int c1 = 16 + lm;
        float shr1 = fmaxf(acc_shr1[r] + shr_b[c1], 0.f);
        float spe1 = fmaxf(acc_spe1[r] + spe_b[c1], 0.f);
        shr_out[(size_t)row * HID + c1] = shr1;
        h_out  [(size_t)row * HID + c1] = spe1 + shr1;
    }
}

// ---------------------------------------------------------------------------
// GAT projection: hbuf[N,16] = x[N,32] @ w[32,16] -- one WMMA per 16-row tile.
// ---------------------------------------------------------------------------
__global__ void gath_kernel(const float* __restrict__ x, const v16h* __restrict__ w_pk,
                            float* __restrict__ hbuf, int N) {
    int wave  = blockIdx.x * (blockDim.x >> 5) + (threadIdx.x >> 5);
    int lane  = threadIdx.x & 31;
    int tiles = (N + 15) >> 4;
    if (wave >= tiles) return;

    int row0 = wave << 4;
    int lm   = lane & 15;
    int koff = (lane < 16) ? 0 : 8;

    int arow = row0 + lm;
    if (arow >= N) arow = N - 1;
    const float* xrow = x + (size_t)arow * HID;

    v16h a;
#pragma unroll
    for (int t = 0; t < 8; ++t) {
        a[t]     = (_Float16)xrow[koff + t];
        a[8 + t] = (_Float16)xrow[16 + koff + t];
    }
    v16h b = w_pk[lane];
    v8f acc = {};
    acc = __builtin_amdgcn_wmma_f32_16x16x32_f16(false, a, false, b,
                                                 (short)0, acc, false, false);

    int rbase = row0 + ((lane < 16) ? 0 : 8);
#pragma unroll
    for (int r = 0; r < 8; ++r) {
        int row = rbase + r;
        if (row < N) hbuf[(size_t)row * GOUT + lm] = acc[r];
    }
}

// ---------------------------------------------------------------------------
// Per-node attention init: as_, ad_, m = e_self (self-loop), s = 0, acc = 0
// ---------------------------------------------------------------------------
__global__ void attn_init_kernel(const float* __restrict__ hbuf,
                                 const float* __restrict__ a_src, const float* __restrict__ a_dst,
                                 float* __restrict__ asb, float* __restrict__ adb,
                                 float* __restrict__ mb, float* __restrict__ sb,
                                 float* __restrict__ accb, int N) {
    int n = blockIdx.x * blockDim.x + threadIdx.x;
    if (n >= N) return;
    const float* h = hbuf + (size_t)n * GOUT;
#pragma unroll
    for (int hd = 0; hd < 4; ++hd) {
        float as = 0.f, ad = 0.f;
#pragma unroll
        for (int o = 0; o < 4; ++o) {
            float v = h[hd * 4 + o];
            as += v * a_src[hd * 4 + o];
            ad += v * a_dst[hd * 4 + o];
        }
        asb[n * 4 + hd] = as;
        adb[n * 4 + hd] = ad;
        mb[n * 4 + hd]  = leakyf(as + ad);   // self-loop always present
        sb[n * 4 + hd]  = 0.f;
    }
#pragma unroll
    for (int k = 0; k < GOUT; ++k) accb[(size_t)n * GOUT + k] = 0.f;
}

__global__ void edge_max_kernel(const int* __restrict__ ej, const int* __restrict__ ei,
                                const float* __restrict__ asb, const float* __restrict__ adb,
                                float* __restrict__ mb, int E) {
    int e = blockIdx.x * blockDim.x + threadIdx.x;
    if (e >= E) return;
    int j = ej[e], i = ei[e];
#pragma unroll
    for (int hd = 0; hd < 4; ++hd) {
        float v = leakyf(asb[j * 4 + hd] + adb[i * 4 + hd]);
        atomicMaxF(&mb[i * 4 + hd], v);
    }
}

__global__ void edge_sum_kernel(const int* __restrict__ ej, const int* __restrict__ ei,
                                const float* __restrict__ asb, const float* __restrict__ adb,
                                const float* __restrict__ mb, float* __restrict__ sb, int E) {
    int e = blockIdx.x * blockDim.x + threadIdx.x;
    if (e >= E) return;
    int j = ej[e], i = ei[e];
#pragma unroll
    for (int hd = 0; hd < 4; ++hd) {
        float v = leakyf(asb[j * 4 + hd] + adb[i * 4 + hd]);
        atomicAdd(&sb[i * 4 + hd], expf(v - mb[i * 4 + hd]));
    }
}

__global__ void self_sum_kernel(const float* __restrict__ asb, const float* __restrict__ adb,
                                const float* __restrict__ mb, float* __restrict__ sb, int N) {
    int n = blockIdx.x * blockDim.x + threadIdx.x;
    if (n >= N) return;
#pragma unroll
    for (int hd = 0; hd < 4; ++hd) {
        float es = leakyf(asb[n * 4 + hd] + adb[n * 4 + hd]);
        sb[n * 4 + hd] += expf(es - mb[n * 4 + hd]);
    }
}

__global__ void edge_acc_kernel(const int* __restrict__ ej, const int* __restrict__ ei,
                                const float* __restrict__ asb, const float* __restrict__ adb,
                                const float* __restrict__ mb, const float* __restrict__ sb,
                                const float* __restrict__ hbuf, float* __restrict__ accb, int E) {
    int e = blockIdx.x * blockDim.x + threadIdx.x;
    if (e >= E) return;
    int j = ej[e], i = ei[e];
#pragma unroll
    for (int hd = 0; hd < 4; ++hd) {
        float v     = leakyf(asb[j * 4 + hd] + adb[i * 4 + hd]);
        float alpha = expf(v - mb[i * 4 + hd]) / (sb[i * 4 + hd] + 1e-16f);
#pragma unroll
        for (int o = 0; o < 4; ++o) {
            int k = hd * 4 + o;
            atomicAdd(&accb[(size_t)i * GOUT + k], alpha * hbuf[(size_t)j * GOUT + k]);
        }
    }
}

// Finalize: self-loop contribution + bias + ReLU + 16->2 predictor
__global__ void finalize_kernel(const float* __restrict__ hbuf,
                                const float* __restrict__ asb, const float* __restrict__ adb,
                                const float* __restrict__ mb, const float* __restrict__ sb,
                                const float* __restrict__ accb,
                                const float* __restrict__ gat_b,
                                const float* __restrict__ pw, const float* __restrict__ pb,
                                float* __restrict__ outp, int N) {
    int n = blockIdx.x * blockDim.x + threadIdx.x;
    if (n >= N) return;
    float xv[GOUT];
#pragma unroll
    for (int hd = 0; hd < 4; ++hd) {
        float es    = leakyf(asb[n * 4 + hd] + adb[n * 4 + hd]);
        float alpha = expf(es - mb[n * 4 + hd]) / (sb[n * 4 + hd] + 1e-16f);
#pragma unroll
        for (int o = 0; o < 4; ++o) {
            int k = hd * 4 + o;
            xv[k] = fmaxf(accb[(size_t)n * GOUT + k] + alpha * hbuf[(size_t)n * GOUT + k]
                          + gat_b[k], 0.f);
        }
    }
#pragma unroll
    for (int c = 0; c < 2; ++c) {
        float y = pb[c];
#pragma unroll
        for (int k = 0; k < GOUT; ++k) y += xv[k] * pw[k * 2 + c];
        outp[(size_t)n * 2 + c] = y;
    }
}

// Discriminator loss for one graph: mean over nodes of
// softplus(sign*disc(shr)) + softplus(sign*disc(h)); block-reduced atomicAdd.
__global__ void disc_kernel(const float* __restrict__ shrb, const float* __restrict__ hb,
                            const float* __restrict__ w1, const float* __restrict__ b1,
                            const float* __restrict__ w2, const float* __restrict__ b2,
                            float sign, float* __restrict__ loss, int N) {
    int n = blockIdx.x * blockDim.x + threadIdx.x;
    float local = 0.f;
    if (n < N) {
#pragma unroll
        for (int which = 0; which < 2; ++which) {
            const float* xp = (which == 0 ? shrb : hb) + (size_t)n * HID;
            float y = b2[0];
            for (int k = 0; k < 16; ++k) {
                float t = b1[k];
#pragma unroll
                for (int d = 0; d < HID; ++d) t += xp[d] * w1[d * 16 + k];
                y += fmaxf(t, 0.f) * w2[k];
            }
            local += softplusf(sign * y);
        }
    }
    __shared__ float red[256];
    red[threadIdx.x] = local;
    __syncthreads();
    for (int s = blockDim.x >> 1; s > 0; s >>= 1) {
        if (threadIdx.x < (unsigned)s) red[threadIdx.x] += red[threadIdx.x + s];
        __syncthreads();
    }
    if (threadIdx.x == 0) atomicAdd(loss, red[0] / (float)N);
}

__global__ void zero_loss_kernel(float* p) {
    if (threadIdx.x == 0 && blockIdx.x == 0) *p = 0.f;
}

// ---------------------------------------------------------------------------
extern "C" void kernel_launch(void* const* d_in, const int* in_sizes, int n_in,
                              void* d_out, int out_size, void* d_ws, size_t ws_size,
                              hipStream_t stream) {
    const float* src      = (const float*)d_in[0];
    const float* trg      = (const float*)d_in[1];
    const int*   src_edge = (const int*)d_in[2];
    const int*   trg_edge = (const int*)d_in[3];
    const float* srcfe_w  = (const float*)d_in[4];
    const float* srcfe_b  = (const float*)d_in[5];
    const float* shrfe_w  = (const float*)d_in[6];
    const float* shrfe_b  = (const float*)d_in[7];
    const float* trgfe_w  = (const float*)d_in[8];
    const float* trgfe_b  = (const float*)d_in[9];
    const float* disc_w1  = (const float*)d_in[10];
    const float* disc_b1  = (const float*)d_in[11];
    const float* disc_w2  = (const float*)d_in[12];
    const float* disc_b2  = (const float*)d_in[13];
    const float* gat_s_w  = (const float*)d_in[14];
    const float* gat_s_as = (const float*)d_in[15];
    const float* gat_s_ad = (const float*)d_in[16];
    const float* gat_s_b  = (const float*)d_in[17];
    const float* pred_s_w = (const float*)d_in[18];
    const float* pred_s_b = (const float*)d_in[19];
    const float* gat_t_w  = (const float*)d_in[20];
    const float* gat_t_as = (const float*)d_in[21];
    const float* gat_t_ad = (const float*)d_in[22];
    const float* gat_t_b  = (const float*)d_in[23];
    const float* pred_t_w = (const float*)d_in[24];
    const float* pred_t_b = (const float*)d_in[25];

    const int N  = in_sizes[0] / FDIM;   // 100000
    const int Nt = in_sizes[1] / FDIM;   // 100000
    const int Es = in_sizes[2] / 2;      // 3200000
    const int Et = in_sizes[3] / 2;

    float* out = (float*)d_out;
    float* loss_slot = out + (size_t)2 * N + (size_t)2 * Nt;

    // Workspace layout (floats, then packed f16 weights)
    float* ws      = (float*)d_ws;
    float* src_h   = ws;
    float* trg_h   = src_h  + (size_t)N  * HID;
    float* src_shr = trg_h  + (size_t)Nt * HID;
    float* trg_shr = src_shr + (size_t)N * HID;
    float* hbuf    = trg_shr + (size_t)Nt * HID;   // reused per graph
    int Nmax = (N > Nt) ? N : Nt;
    float* asb  = hbuf + (size_t)Nmax * GOUT;
    float* adb  = asb  + (size_t)Nmax * 4;
    float* mb   = adb  + (size_t)Nmax * 4;
    float* sb   = mb   + (size_t)Nmax * 4;
    float* accb = sb   + (size_t)Nmax * 4;
    // packed f16 weight fragments (32B-aligned: all prior extents are x16 floats)
    _Float16* pk_base  = (_Float16*)(accb + (size_t)Nmax * GOUT);
    const size_t FEAT_FRAGS = 4 * 2 * 32 * 16;   // kt=4, nt=2, 32 lanes, 16 halfs
    const size_t GAT_FRAGS  = 1 * 1 * 32 * 16;
    _Float16* pk_srcfe = pk_base;
    _Float16* pk_shrfe = pk_srcfe + FEAT_FRAGS;
    _Float16* pk_trgfe = pk_shrfe + FEAT_FRAGS;
    _Float16* pk_gat_s = pk_trgfe + FEAT_FRAGS;
    _Float16* pk_gat_t = pk_gat_s + GAT_FRAGS;

    const int TB = 256;
    dim3 nodeGrid((N + TB - 1) / TB);
    dim3 nodeGridT((Nt + TB - 1) / TB);

    // feature GEMMs: 1 wave per 16-row tile, 4 waves per block
    int tilesS = (N + 15) / 16, tilesT = (Nt + 15) / 16;
    dim3 featGridS((tilesS + 3) / 4), featGridT((tilesT + 3) / 4);

    zero_loss_kernel<<<1, 32, 0, stream>>>(loss_slot);

    // Pack weights into WMMA fragment layout (once per call; trivial cost)
    pack_w_kernel<<<1, 256, 0, stream>>>(srcfe_w, pk_srcfe, 4, 2, HID);
    pack_w_kernel<<<1, 256, 0, stream>>>(shrfe_w, pk_shrfe, 4, 2, HID);
    pack_w_kernel<<<1, 256, 0, stream>>>(trgfe_w, pk_trgfe, 4, 2, HID);
    pack_w_kernel<<<1, 32, 0, stream>>>(gat_s_w, pk_gat_s, 1, 1, GOUT);
    pack_w_kernel<<<1, 32, 0, stream>>>(gat_t_w, pk_gat_t, 1, 1, GOUT);

    feat_kernel<<<featGridS, 128, 0, stream>>>(src, (const v16h*)pk_srcfe, srcfe_b,
                                               (const v16h*)pk_shrfe, shrfe_b,
                                               src_h, src_shr, N);
    feat_kernel<<<featGridT, 128, 0, stream>>>(trg, (const v16h*)pk_trgfe, trgfe_b,
                                               (const v16h*)pk_shrfe, shrfe_b,
                                               trg_h, trg_shr, Nt);

    disc_kernel<<<nodeGrid, TB, 0, stream>>>(src_shr, src_h, disc_w1, disc_b1,
                                             disc_w2, disc_b2, 1.f, loss_slot, N);
    disc_kernel<<<nodeGridT, TB, 0, stream>>>(trg_shr, trg_h, disc_w1, disc_b1,
                                              disc_w2, disc_b2, -1.f, loss_slot, Nt);

    // ---- src graph GAT ----
    {
        const int* ej = src_edge;
        const int* ei = src_edge + Es;
        dim3 edgeGrid((Es + TB - 1) / TB);
        gath_kernel<<<featGridS, 128, 0, stream>>>(src_h, (const v16h*)pk_gat_s, hbuf, N);
        attn_init_kernel<<<nodeGrid, TB, 0, stream>>>(hbuf, gat_s_as, gat_s_ad,
                                                      asb, adb, mb, sb, accb, N);
        edge_max_kernel<<<edgeGrid, TB, 0, stream>>>(ej, ei, asb, adb, mb, Es);
        edge_sum_kernel<<<edgeGrid, TB, 0, stream>>>(ej, ei, asb, adb, mb, sb, Es);
        self_sum_kernel<<<nodeGrid, TB, 0, stream>>>(asb, adb, mb, sb, N);
        edge_acc_kernel<<<edgeGrid, TB, 0, stream>>>(ej, ei, asb, adb, mb, sb, hbuf, accb, Es);
        finalize_kernel<<<nodeGrid, TB, 0, stream>>>(hbuf, asb, adb, mb, sb, accb,
                                                     gat_s_b, pred_s_w, pred_s_b, out, N);
    }
    // ---- trg graph GAT (reuses scratch; stream-ordered) ----
    {
        const int* ej = trg_edge;
        const int* ei = trg_edge + Et;
        dim3 edgeGrid((Et + TB - 1) / TB);
        float* outT = out + (size_t)2 * N;
        gath_kernel<<<featGridT, 128, 0, stream>>>(trg_h, (const v16h*)pk_gat_t, hbuf, Nt);
        attn_init_kernel<<<nodeGridT, TB, 0, stream>>>(hbuf, gat_t_as, gat_t_ad,
                                                       asb, adb, mb, sb, accb, Nt);
        edge_max_kernel<<<edgeGrid, TB, 0, stream>>>(ej, ei, asb, adb, mb, Et);
        edge_sum_kernel<<<edgeGrid, TB, 0, stream>>>(ej, ei, asb, adb, mb, sb, Et);
        self_sum_kernel<<<nodeGridT, TB, 0, stream>>>(asb, adb, mb, sb, Nt);
        edge_acc_kernel<<<edgeGrid, TB, 0, stream>>>(ej, ei, asb, adb, mb, sb, hbuf, accb, Et);
        finalize_kernel<<<nodeGridT, TB, 0, stream>>>(hbuf, asb, adb, mb, sb, accb,
                                                      gat_t_b, pred_t_w, pred_t_b, outT, Nt);
    }
    (void)ws_size; (void)n_in; (void)out_size;
}
